// E_LoFTR_Fuse_609885356633
// MI455X (gfx1250) — compile-verified, matrix operand order checked by reading, not run on hardware
//
#include <hip/hip_runtime.h>
#include <hip/hip_bf16.h>
#include <math.h>

// Problem constants from the reference (h0c=w0c=h1c=w1c=64, C=256, N=4)
#define NB   4
#define LQ   4096
#define SQ   4096
#define CC   256
#define HC   64
#define WC   64
#define BORDER 2
#define THRV   0.2f
#define SIMSCALE (1.0f / (256.0f * 0.1f))   // 1/(C*TEMPERATURE): includes both 1/sqrt(C) factors

typedef __attribute__((ext_vector_type(16))) __bf16 v16bf;
typedef __attribute__((ext_vector_type(8)))  float  v8f;

union Frag16 { uint4 q[2]; v16bf v; };

__device__ __forceinline__ unsigned short f2bf(float f) {
    unsigned u = __float_as_uint(f);
    u += 0x7FFFu + ((u >> 16) & 1u);      // round-to-nearest-even
    return (unsigned short)(u >> 16);
}

#define AST 264   // padded LDS row stride in bf16 elements (256 + 8)

// ---------------------------------------------------------------------------
// K1: sim[n,i,j] = (feat0[n,i,:] . feat1[n,j,:]) * SIMSCALE  via bf16 WMMA
// block = 256 threads (8 waves) -> 64x32 output tile
// grid  = (SQ/32, LQ/64, NB)
// ---------------------------------------------------------------------------
__global__ __launch_bounds__(256)
void k_gemm_sim(const float* __restrict__ f0, const float* __restrict__ f1,
                float* __restrict__ sim)
{
    __shared__ __align__(16) unsigned short As[64 * AST];
    __shared__ __align__(16) unsigned short Bs[32 * AST];

    const int n  = blockIdx.z;
    const int i0 = blockIdx.y * 64;
    const int j0 = blockIdx.x * 32;

    // cooperative global->LDS load with fp32 -> bf16 conversion
    const float4* ga = (const float4*)(f0 + ((size_t)n * LQ + i0) * CC);
    #pragma unroll
    for (int it = 0; it < 16; ++it) {
        int f4 = threadIdx.x + it * 256;         // 0..4095 float4s (64 rows x 64)
        int r  = f4 >> 6;
        int c4 = f4 & 63;
        float4 x = ga[r * (CC / 4) + c4];
        unsigned p0 = (unsigned)f2bf(x.x) | ((unsigned)f2bf(x.y) << 16);
        unsigned p1 = (unsigned)f2bf(x.z) | ((unsigned)f2bf(x.w) << 16);
        *(uint2*)&As[r * AST + c4 * 4] = make_uint2(p0, p1);
    }
    const float4* gb = (const float4*)(f1 + ((size_t)n * SQ + j0) * CC);
    #pragma unroll
    for (int it = 0; it < 8; ++it) {
        int f4 = threadIdx.x + it * 256;         // 0..2047 float4s (32 rows x 64)
        int r  = f4 >> 6;
        int c4 = f4 & 63;
        float4 x = gb[r * (CC / 4) + c4];
        unsigned p0 = (unsigned)f2bf(x.x) | ((unsigned)f2bf(x.y) << 16);
        unsigned p1 = (unsigned)f2bf(x.z) | ((unsigned)f2bf(x.w) << 16);
        *(uint2*)&Bs[r * AST + c4 * 4] = make_uint2(p0, p1);
    }
    __syncthreads();

    const int wave = threadIdx.x >> 5;
    const int lane = threadIdx.x & 31;
    const int wm   = wave & 3;    // 4 row-tiles of 16
    const int wn   = wave >> 2;   // 2 col-tiles of 16
    const int mr   = lane & 15;   // A row / B col within tile
    const int kh   = lane >> 4;   // K half selector (ISA per-lane layout)

    v8f acc = {};
    const unsigned short* ap = &As[(wm * 16 + mr) * AST + kh * 8];
    const unsigned short* bp = &Bs[(wn * 16 + mr) * AST + kh * 8];
    #pragma unroll
    for (int k = 0; k < 8; ++k) {               // K = 8 * 32 = 256
        Frag16 a, b;
        a.q[0] = *(const uint4*)(ap + k * 32);        // elems 0..7  : K = k*32 + kh*8 + e
        a.q[1] = *(const uint4*)(ap + k * 32 + 16);   // elems 8..15 : K = k*32 + 16 + kh*8 + e
        b.q[0] = *(const uint4*)(bp + k * 32);
        b.q[1] = *(const uint4*)(bp + k * 32 + 16);
        acc = __builtin_amdgcn_wmma_f32_16x16x32_bf16(false, a.v, false, b.v,
                                                      (short)0, acc, false, false);
    }

    // D layout: VGPR r -> M = r + 8*(lane>=16), N = lane%16
    const int gj = j0 + wn * 16 + mr;
    float* orow = sim + ((size_t)n * LQ + (i0 + wm * 16 + kh * 8)) * SQ + gj;
    #pragma unroll
    for (int r = 0; r < 8; ++r)
        orow[(size_t)r * SQ] = acc[r] * SIMSCALE;
}

// ---------------------------------------------------------------------------
// K2: per-row online log-sum-exp: rowOff[n,i] = max_j + log(sum_j exp(x - max))
// ---------------------------------------------------------------------------
__global__ __launch_bounds__(256)
void k_row_stats(const float* __restrict__ sim, float* __restrict__ rowOff)
{
    __shared__ float sm[256], ss[256];
    const int i = blockIdx.x, n = blockIdx.y;
    const float* row = sim + ((size_t)n * LQ + i) * SQ;
    float m = -INFINITY, s = 0.f;
    for (int j = threadIdx.x; j < SQ; j += 256) {
        float x = row[j];
        if (x > m) { s = s * __expf(m - x) + 1.f; m = x; }
        else       { s += __expf(x - m); }
    }
    sm[threadIdx.x] = m; ss[threadIdx.x] = s;
    __syncthreads();
    for (int st = 128; st > 0; st >>= 1) {
        if (threadIdx.x < st) {
            float m1 = sm[threadIdx.x], s1 = ss[threadIdx.x];
            float m2 = sm[threadIdx.x + st], s2 = ss[threadIdx.x + st];
            float M = fmaxf(m1, m2);
            sm[threadIdx.x] = M;
            ss[threadIdx.x] = s1 * __expf(m1 - M) + s2 * __expf(m2 - M);
        }
        __syncthreads();
    }
    if (threadIdx.x == 0) rowOff[n * LQ + i] = sm[0] + __logf(ss[0]);
}

// ---------------------------------------------------------------------------
// K3: per-column online log-sum-exp (coalesced: thread owns one column)
// ---------------------------------------------------------------------------
__global__ __launch_bounds__(256)
void k_col_stats(const float* __restrict__ sim, float* __restrict__ colOff)
{
    const int j = blockIdx.x * 256 + threadIdx.x;
    const int n = blockIdx.y;
    const float* base = sim + (size_t)n * LQ * SQ + j;
    float m = -INFINITY, s = 0.f;
    for (int i = 0; i < LQ; ++i) {
        float x = base[(size_t)i * SQ];
        if (x > m) { s = s * __expf(m - x) + 1.f; m = x; }
        else       { s += __expf(x - m); }
    }
    colOff[n * SQ + j] = m + __logf(s);
}

// ---------------------------------------------------------------------------
// K4: conf = exp(2*sim - rowOff_i - colOff_j) in place; fused per-row
//     max + first-argmax (tie -> smaller j, matching argmax semantics)
// ---------------------------------------------------------------------------
__global__ __launch_bounds__(256)
void k_transform_rowmax(float* __restrict__ conf,
                        const float* __restrict__ rowOff,
                        const float* __restrict__ colOff,
                        float* __restrict__ rowMax, int* __restrict__ rowArg)
{
    __shared__ float sv[256];
    __shared__ int   si[256];
    const int i = blockIdx.x, n = blockIdx.y;
    const float ro = rowOff[n * LQ + i];
    float* row = conf + ((size_t)n * LQ + i) * SQ;
    const float* co = colOff + (size_t)n * SQ;
    float best = -1.f; int barg = 0;
    for (int j = threadIdx.x; j < SQ; j += 256) {   // ascending j per thread
        float c = __expf(2.f * row[j] - ro - co[j]);
        row[j] = c;
        if (c > best) { best = c; barg = j; }
    }
    sv[threadIdx.x] = best; si[threadIdx.x] = barg;
    __syncthreads();
    for (int st = 128; st > 0; st >>= 1) {
        if (threadIdx.x < st) {
            float v2 = sv[threadIdx.x + st]; int i2 = si[threadIdx.x + st];
            if (v2 > sv[threadIdx.x] ||
                (v2 == sv[threadIdx.x] && i2 < si[threadIdx.x])) {
                sv[threadIdx.x] = v2; si[threadIdx.x] = i2;
            }
        }
        __syncthreads();
    }
    if (threadIdx.x == 0) {
        rowMax[n * LQ + i] = sv[0];
        rowArg[n * LQ + i] = si[0];
    }
}

// ---------------------------------------------------------------------------
// K5: per-column max of conf (value only; needed for mutual-top-1 test)
// ---------------------------------------------------------------------------
__global__ __launch_bounds__(256)
void k_col_max(const float* __restrict__ conf, float* __restrict__ colMax)
{
    const int j = blockIdx.x * 256 + threadIdx.x;
    const int n = blockIdx.y;
    const float* base = conf + (size_t)n * LQ * SQ + j;
    float m = -1.f;
    for (int i = 0; i < LQ; ++i) m = fmaxf(m, base[(size_t)i * SQ]);
    colMax[n * SQ + j] = m;
}

// ---------------------------------------------------------------------------
// K6: threshold + border + mutual-top-1 -> mask_v, j_ids, mconf
// ---------------------------------------------------------------------------
__device__ __forceinline__ bool border_ok(int idx) {
    int h = idx >> 6, w = idx & 63;   // 64x64 grid
    return (h >= BORDER) && (h < HC - BORDER) && (w >= BORDER) && (w < WC - BORDER);
}

__global__ __launch_bounds__(256)
void k_finalize(const float* __restrict__ rowMax, const int* __restrict__ rowArg,
                const float* __restrict__ colMax,
                float* __restrict__ maskv, float* __restrict__ jids,
                float* __restrict__ mconf)
{
    const int idx = blockIdx.x * 256 + threadIdx.x;   // over NB*LQ
    if (idx >= NB * LQ) return;
    const int n = idx / LQ;
    const int i = idx - n * LQ;
    const int j = rowArg[idx];
    const float v = rowMax[idx];
    bool valid = (v > THRV) && border_ok(i) && border_ok(j) &&
                 (v == colMax[(size_t)n * SQ + j]);
    maskv[idx] = valid ? 1.f : 0.f;
    jids[idx]  = (float)(valid ? j : 0);
    mconf[idx] = valid ? v : 0.f;
}

// ---------------------------------------------------------------------------
extern "C" void kernel_launch(void* const* d_in, const int* in_sizes, int n_in,
                              void* d_out, int out_size, void* d_ws, size_t ws_size,
                              hipStream_t stream)
{
    const float* f0 = (const float*)d_in[0];
    const float* f1 = (const float*)d_in[1];

    // outputs concatenated flat: conf [N,L,S], mask_v [N,L], j_ids [N,L], mconf [N,L]
    float* out   = (float*)d_out;
    float* conf  = out;
    float* maskv = out + (size_t)NB * LQ * SQ;
    float* jids  = maskv + (size_t)NB * LQ;
    float* mconf = jids  + (size_t)NB * LQ;

    // scratch layout (320 KB total)
    float* rowOff = (float*)d_ws;
    float* colOff = rowOff + NB * LQ;
    float* rowMax = colOff + NB * SQ;
    float* colMax = rowMax + NB * LQ;
    int*   rowArg = (int*)(colMax + NB * SQ);

    k_gemm_sim        <<<dim3(SQ / 32, LQ / 64, NB), 256, 0, stream>>>(f0, f1, conf);
    k_row_stats       <<<dim3(LQ, NB),               256, 0, stream>>>(conf, rowOff);
    k_col_stats       <<<dim3(SQ / 256, NB),         256, 0, stream>>>(conf, colOff);
    k_transform_rowmax<<<dim3(LQ, NB),               256, 0, stream>>>(conf, rowOff, colOff, rowMax, rowArg);
    k_col_max         <<<dim3(SQ / 256, NB),         256, 0, stream>>>(conf, colMax);
    k_finalize        <<<dim3((NB * LQ) / 256),      256, 0, stream>>>(rowMax, rowArg, colMax, maskv, jids, mconf);
}
